// attentiona_35390530519834
// MI455X (gfx1250) — matrix operand order repaired and checked by reading, not compiled, not run on hardware
//
#include <hip/hip_runtime.h>
#include <hip/hip_bf16.h>

// ---- problem constants (B = 1) ----
#define kT  2048   // sequence length
#define kD  1024   // model dims
#define kH  16     // heads
#define kHD 64     // head dim

typedef __attribute__((ext_vector_type(8)))  __bf16 v8bf;
typedef __attribute__((ext_vector_type(16))) __bf16 v16bf;
typedef __attribute__((ext_vector_type(8)))  float  v8f;

static __device__ __forceinline__ __bf16 f2bf(float f) {
  unsigned u = __builtin_bit_cast(unsigned, f);
  u += 0x7FFFu + ((u >> 16) & 1u);           // round-to-nearest-even
  unsigned short h = (unsigned short)(u >> 16);
  return __builtin_bit_cast(__bf16, h);
}

static __device__ __forceinline__ v8bf ldv8(const __bf16* p) {
  return *(const v8bf*)p;                     // 16B aligned by construction
}

static __device__ __forceinline__ v16bf pack16(v8bf lo, v8bf hi) {
  v16bf r;
#pragma unroll
  for (int i = 0; i < 8; ++i) { r[i] = lo[i]; r[i + 8] = hi[i]; }
  return r;
}

static __device__ __forceinline__ v8f wmma_bf16(v16bf a, v16bf b, v8f c) {
  return __builtin_amdgcn_wmma_f32_16x16x32_bf16(
      /*neg_a=*/false, a, /*neg_b=*/false, b,
      /*c_mod=*/(short)0, c, /*reuse_a=*/false, /*reuse_b=*/false);
}

// ---------------- LayerNorm -> bf16 h ----------------
__global__ void ln_to_bf16(const float* __restrict__ x,
                           const float* __restrict__ w,
                           const float* __restrict__ b,
                           __bf16* __restrict__ hb) {
  const int row = blockIdx.x;
  const float* xr = x + (size_t)row * kD;
  float s = 0.f, s2 = 0.f;
  for (int i = threadIdx.x; i < kD; i += 256) {
    float v = xr[i]; s += v; s2 += v * v;
  }
#pragma unroll
  for (int off = 16; off; off >>= 1) {
    s  += __shfl_xor(s,  off);
    s2 += __shfl_xor(s2, off);
  }
  __shared__ float rs[8], rq[8];
  const int wave = threadIdx.x >> 5;
  if ((threadIdx.x & 31) == 0) { rs[wave] = s; rq[wave] = s2; }
  __syncthreads();
  float ts = 0.f, tq = 0.f;
#pragma unroll
  for (int i = 0; i < 8; ++i) { ts += rs[i]; tq += rq[i]; }
  const float mu   = ts * (1.f / kD);
  const float var  = tq * (1.f / kD) - mu * mu;
  const float rstd = rsqrtf(var + 1e-5f);
  for (int i = threadIdx.x; i < kD; i += 256)
    hb[(size_t)row * kD + i] = f2bf((xr[i] - mu) * rstd * w[i] + b[i]);
}

// ---- weight fp32 [K][N] -> bf16 transposed [N][K] (WMMA B-operand layout) ----
__global__ void w_to_bf16_T(const float* __restrict__ w, __bf16* __restrict__ wT,
                            int K, int N) {
  int tid = blockIdx.x * 256 + threadIdx.x;
  if (tid >= K * N) return;
  int n = tid / K, k = tid - n * K;
  wT[tid] = f2bf(w[(size_t)k * N + n]);
}

// ---- register-blocked WMMA GEMM: one wave computes a 16(M) x 64(N) tile ------
// C = A[M,K] * Bt[N,K]^T ; A-tile loads reused across 4 N-tiles (4 wmma / k-step)
__global__ void gemm_wmma(const __bf16* __restrict__ A,
                          const __bf16* __restrict__ Bt,
                          float* __restrict__ C, int N, int K) {
  const int tilesN = N >> 6;                       // 64-wide N blocks
  const int m0 = (blockIdx.x / tilesN) << 4;
  const int n0 = (blockIdx.x % tilesN) << 6;
  const int lane = threadIdx.x;
  const int half = lane >> 4;
  const int l15  = lane & 15;

  const __bf16* arow = A + (size_t)(m0 + l15) * K;       // A: lane = M row
  const __bf16* brow[4];
#pragma unroll
  for (int nt = 0; nt < 4; ++nt)
    brow[nt] = Bt + (size_t)(n0 + nt * 16 + l15) * K;    // B: lane = N col

  v8f acc[4] = {};
  for (int k0 = 0; k0 < K; k0 += 32) {
    // A operand, contiguous k order (matches B): elem[e] <-> k = k0 + half*16 + e
    v16bf av = pack16(ldv8(arow + k0 + half * 16),
                      ldv8(arow + k0 + half * 16 + 8));
#pragma unroll
    for (int nt = 0; nt < 4; ++nt) {
      // B operand, contiguous k order (matches A): elem[e] <-> k = k0 + half*16 + e
      v16bf bv = pack16(ldv8(brow[nt] + k0 + half * 16),
                        ldv8(brow[nt] + k0 + half * 16 + 8));
      acc[nt] = wmma_bf16(av, bv, acc[nt]);
    }
    if (k0 + 256 < K) {
      __builtin_prefetch(arow + k0 + 256, 0, 3);         // global_prefetch_b8
      __builtin_prefetch(brow[0] + k0 + 256, 0, 3);
      __builtin_prefetch(brow[2] + k0 + 256, 0, 3);
    }
  }
#pragma unroll
  for (int nt = 0; nt < 4; ++nt)
#pragma unroll
    for (int r = 0; r < 8; ++r) {                        // D layout: row = r + 8*half
      const int row = m0 + r + (half << 3);
      const int col = n0 + nt * 16 + l15;
      C[(size_t)row * N + col] = acc[nt][r];
    }
}

// ---------------- RoPE + head split (q,k rotated; v transposed) ----------------
__global__ void rope_split(const float* __restrict__ qf,
                           const float* __restrict__ kvf,
                           const float* __restrict__ theta_p,
                           __bf16* __restrict__ qb,
                           __bf16* __restrict__ kb,
                           __bf16* __restrict__ vT) {
  const int tid  = blockIdx.x * 256 + threadIdx.x;  // kT*kH*32 threads
  const int f    = tid & 31;
  const int head = (tid >> 5) & 15;
  const int t    = tid >> 9;
  const float theta = theta_p[0];
  // freqs_base[f] = 0.2*(21^(f/31)-1); freq = theta/220 * base
  const float base = __expf((float)f * (logf(21.f) * (1.f / 31.f))) - 1.f;
  const float freq = theta * (1.f / 1100.f) * base;
  float sn, cs;
  __sincosf((float)t * freq, &sn, &cs);

  { // q
    const int src = t * kD + head * kHD + 2 * f;
    const float a = qf[src], b = qf[src + 1];
    const size_t dst = ((size_t)(head * kT + t)) * kHD + 2 * f;
    qb[dst]     = f2bf(a * cs - b * sn);
    qb[dst + 1] = f2bf(a * sn + b * cs);
  }
  { // k (first half of kv)
    const int src = t * (2 * kD) + head * kHD + 2 * f;
    const float a = kvf[src], b = kvf[src + 1];
    const size_t dst = ((size_t)(head * kT + t)) * kHD + 2 * f;
    kb[dst]     = f2bf(a * cs - b * sn);
    kb[dst + 1] = f2bf(a * sn + b * cs);
  }
  { // v (second half) -> transposed [head][hd][T]
    const int src = t * (2 * kD) + kD + head * kHD + 2 * f;
    const int d0  = head * kHD + 2 * f;
    vT[(size_t)d0 * kT + t]       = f2bf(kvf[src]);
    vT[(size_t)(d0 + 1) * kT + t] = f2bf(kvf[src + 1]);
  }
}

// ---- streaming Taylor-softmax attention: one wave per (head, 32 query rows) ---
// Two 16-row i-tiles share every K and V B-operand -> 16 wmma per 32-column step.
__global__ void attn_wmma(const __bf16* __restrict__ qb,
                          const __bf16* __restrict__ kb,
                          const __bf16* __restrict__ vT,
                          __bf16* __restrict__ wvb) {
  const int head = blockIdx.x >> 6;          // kT/32 = 64 row-blocks per head
  const int i0   = (blockIdx.x & 63) << 5;   // 32 query rows per wave
  const int lane = threadIdx.x;
  const int half = lane >> 4;
  const int l15  = lane & 15;

  __shared__ __bf16 lds_p[2][16][32];        // P staging (D-layout -> A-layout)

  // Q tiles (2 x 16 rows x 64), two K=32 A-chunks each, reused for all j
  v16bf Aq[2][2];
#pragma unroll
  for (int it = 0; it < 2; ++it) {
    const __bf16* qrow =
        qb + ((size_t)(head * kT) + i0 + it * 16 + l15) * kHD;
#pragma unroll
    for (int c = 0; c < 2; ++c)
      Aq[it][c] = pack16(ldv8(qrow + c * 32 + half * 16),
                         ldv8(qrow + c * 32 + half * 16 + 8));
  }

  v8f zacc = {};
  v8f accWV[2][4];                           // 2 x (16 x 64) outputs
#pragma unroll
  for (int it = 0; it < 2; ++it)
#pragma unroll
    for (int dc = 0; dc < 4; ++dc) accWV[it][dc] = zacc;
  float dsum[2][8];
#pragma unroll
  for (int it = 0; it < 2; ++it)
#pragma unroll
    for (int r = 0; r < 8; ++r) dsum[it][r] = 0.f;

  const float scale = 0.125f;                // 64^-0.5

  for (int j0 = 0; j0 < kT; j0 += 32) {
    // ---- K B-operands for 32 columns, loaded once, used by both i-tiles ----
    v16bf Bk[2][2];
#pragma unroll
    for (int jt = 0; jt < 2; ++jt) {
      const __bf16* krow =
          kb + ((size_t)(head * kT) + j0 + jt * 16 + l15) * kHD;
#pragma unroll
      for (int c = 0; c < 2; ++c)
        Bk[jt][c] = pack16(ldv8(krow + c * 32 + half * 16),
                           ldv8(krow + c * 32 + half * 16 + 8));
    }
    // ---- S = Q K^T : 8 wmma ----
    v8f st[2][2];
#pragma unroll
    for (int it = 0; it < 2; ++it)
#pragma unroll
      for (int jt = 0; jt < 2; ++jt) {
        v8f a = zacc;
        a = wmma_bf16(Aq[it][0], Bk[jt][0], a);
        a = wmma_bf16(Aq[it][1], Bk[jt][1], a);
        st[it][jt] = a;
      }
    __syncthreads();                         // previous iter's LDS reads done
    // ---- t = 1 + x + x^2/2 ; stage into LDS ; accumulate row sums ----
#pragma unroll
    for (int it = 0; it < 2; ++it)
#pragma unroll
      for (int jt = 0; jt < 2; ++jt)
#pragma unroll
        for (int r = 0; r < 8; ++r) {
          const float x  = st[it][jt][r] * scale;
          const float tv = 1.f + x + 0.5f * x * x;
          dsum[it][r] += tv;
          lds_p[it][r + (half << 3)][jt * 16 + l15] = f2bf(tv);
        }
    __syncthreads();
    // ---- reload P tiles in A-layout (16x32) ----
    v16bf Ap[2];
#pragma unroll
    for (int it = 0; it < 2; ++it)
#pragma unroll
      for (int g = 0; g < 2; ++g)
#pragma unroll
        for (int j2 = 0; j2 < 8; ++j2)
          Ap[it][g * 8 + j2] = lds_p[it][l15][half * 16 + g * 8 + j2];
    // ---- accWV += P * V : V B-operands loaded once, 8 wmma ----
#pragma unroll
    for (int dc = 0; dc < 4; ++dc) {
      const __bf16* vrow =
          vT + ((size_t)(head * kHD) + dc * 16 + l15) * kT + j0 + half * 16;
      v16bf bv = pack16(ldv8(vrow), ldv8(vrow + 8));
#pragma unroll
      for (int it = 0; it < 2; ++it)
        accWV[it][dc] = wmma_bf16(Ap[it], bv, accWV[it][dc]);
    }
  }

  // row-sum reduction within each 16-lane half (rows r+8*half)
#pragma unroll
  for (int it = 0; it < 2; ++it)
#pragma unroll
    for (int r = 0; r < 8; ++r) {
      float v = dsum[it][r];
      v += __shfl_xor(v, 1);
      v += __shfl_xor(v, 2);
      v += __shfl_xor(v, 4);
      v += __shfl_xor(v, 8);
      dsum[it][r] = v;
    }
  // normalize and store wv as bf16 [T][D]
#pragma unroll
  for (int it = 0; it < 2; ++it)
#pragma unroll
    for (int dc = 0; dc < 4; ++dc)
#pragma unroll
      for (int r = 0; r < 8; ++r) {
        const float v  = accWV[it][dc][r] / dsum[it][r];
        const int row  = i0 + it * 16 + r + (half << 3);
        const int col  = head * kHD + dc * 16 + l15;
        wvb[(size_t)row * kD + col] = f2bf(v);
      }
}

extern "C" void kernel_launch(void* const* d_in, const int* in_sizes, int n_in,
                              void* d_out, int out_size, void* d_ws, size_t ws_size,
                              hipStream_t stream) {
  const float* x     = (const float*)d_in[0];
  const float* ln_w  = (const float*)d_in[1];
  const float* ln_b  = (const float*)d_in[2];
  const float* w_q   = (const float*)d_in[3];
  const float* w_kv  = (const float*)d_in[4];
  const float* w_o   = (const float*)d_in[5];
  const float* theta = (const float*)d_in[6];
  (void)in_sizes; (void)n_in; (void)out_size; (void)ws_size;

  char* ws = (char*)d_ws;
  size_t off = 0;
  auto alloc = [&](size_t bytes) {
    char* p = ws + off;
    off += (bytes + 255) & ~(size_t)255;
    return p;
  };
  __bf16* hb   = (__bf16*)alloc((size_t)kT * kD * 2);        //  4 MB
  __bf16* wqT  = (__bf16*)alloc((size_t)kD * kD * 2);        //  2 MB
  __bf16* wkvT = (__bf16*)alloc((size_t)(2 * kD) * kD * 2);  //  4 MB
  __bf16* woT  = (__bf16*)alloc((size_t)kD * kD * 2);        //  2 MB
  float*  qf   = (float*) alloc((size_t)kT * kD * 4);        //  8 MB
  float*  kvf  = (float*) alloc((size_t)kT * 2 * kD * 4);    // 16 MB
  __bf16* qb   = (__bf16*)alloc((size_t)kH * kT * kHD * 2);  //  4 MB
  __bf16* kb   = (__bf16*)alloc((size_t)kH * kT * kHD * 2);  //  4 MB
  __bf16* vT   = (__bf16*)alloc((size_t)kH * kHD * kT * 2);  //  4 MB
  __bf16* wvb  = (__bf16*)alloc((size_t)kT * kD * 2);        //  4 MB

  ln_to_bf16<<<kT, 256, 0, stream>>>(x, ln_w, ln_b, hb);
  w_to_bf16_T<<<(kD * kD) / 256, 256, 0, stream>>>(w_q, wqT, kD, kD);
  w_to_bf16_T<<<(kD * 2 * kD) / 256, 256, 0, stream>>>(w_kv, wkvT, kD, 2 * kD);
  w_to_bf16_T<<<(kD * kD) / 256, 256, 0, stream>>>(w_o, woT, kD, kD);

  gemm_wmma<<<(kT / 16) * (kD / 64), 32, 0, stream>>>(hb, wqT, qf, kD, kD);
  gemm_wmma<<<(kT / 16) * ((2 * kD) / 64), 32, 0, stream>>>(hb, wkvT, kvf, 2 * kD, kD);

  rope_split<<<(kT * kH * 32) / 256, 256, 0, stream>>>(qf, kvf, theta, qb, kb, vT);

  attn_wmma<<<kH * (kT / 32), 32, 0, stream>>>(qb, kb, vT, wvb);

  gemm_wmma<<<(kT / 16) * (kD / 64), 32, 0, stream>>>(wvb, woT, (float*)d_out, kD, kD);
}
